// ImprovedMessagePassingLayer_77601469104426
// MI455X (gfx1250) — matrix operand order repaired and checked by reading, not compiled, not run on hardware
//
#include <hip/hip_runtime.h>
#include <hip/hip_bf16.h>
#include <stdint.h>

#define Bdim 32
#define Ndim 1024
#define Hdim 128
#define EPS 1e-5f

typedef __attribute__((ext_vector_type(16))) __bf16 v16bf;
typedef __attribute__((ext_vector_type(8)))  float  v8f;

union Frag16 { v16bf v; uint4 q[2]; };

__device__ __forceinline__ unsigned short f2bf(float f) {
    unsigned int u = __float_as_uint(f);
    u += 0x7FFFu + ((u >> 16) & 1u);   // round-to-nearest-even
    return (unsigned short)(u >> 16);
}

// A fragment: 16x32 bf16 tile from row-major matrix (ld in elements).
// Lanes 0-15: row=lane, elems = K[0..7], K[16..23]; lanes 16-31: row=lane-16, K[8..15], K[24..31].
__device__ __forceinline__ v16bf load_a_frag(const unsigned short* src, int ld, int lane) {
    Frag16 f;
    int row = lane & 15;
    int kb  = (lane < 16) ? 0 : 8;
    const unsigned short* p = src + (size_t)row * ld + kb;
    f.q[0] = *(const uint4*)(p);
    f.q[1] = *(const uint4*)(p + 16);
    return f.v;
}

// B fragment: 32x16 bf16 tile, loaded from BT (N x K row-major transpose of B).
// Lanes 0-15: col=lane, K[0..7], K[16..23]; lanes 16-31: col=lane-16, K[8..15], K[24..31].
__device__ __forceinline__ v16bf load_bt_frag(const unsigned short* src, int ld, int lane) {
    Frag16 f;
    int col = lane & 15;
    int kb  = (lane < 16) ? 0 : 8;
    const unsigned short* p = src + (size_t)col * ld + kb;
    f.q[0] = *(const uint4*)(p);
    f.q[1] = *(const uint4*)(p + 16);
    return f.v;
}

__device__ __forceinline__ v8f wmma_bf16(v16bf a, v16bf b, v8f c) {
    return __builtin_amdgcn_wmma_f32_16x16x32_bf16(false, a, false, b, (short)0, c, false, false);
}

// ---------------- K0a: mask -> bf16 (row-major + transposed) + degree -------
__global__ void k_prep_mask(const int* __restrict__ adj,
                            unsigned short* __restrict__ mask_bf,
                            unsigned short* __restrict__ maskT_bf,
                            float* __restrict__ deg) {
    __shared__ float sred[256];
    int j = blockIdx.x;
    float d = 0.f;
    for (int i = threadIdx.x; i < Ndim; i += 256) {
        float m = (adj[(size_t)j * Ndim + i] > 0) ? 1.0f : 0.0f;
        unsigned short mb = f2bf(m);
        mask_bf[(size_t)j * Ndim + i]  = mb;
        maskT_bf[(size_t)i * Ndim + j] = mb;
        d += m;
    }
    sred[threadIdx.x] = d;
    __syncthreads();
    for (int s = 128; s > 0; s >>= 1) {
        if (threadIdx.x < s) sred[threadIdx.x] += sred[threadIdx.x + s];
        __syncthreads();
    }
    if (threadIdx.x == 0) deg[j] = sred[0];
}

// ---------------- K0b: node f32 -> bf16 row-major + bf16 transposed ---------
__global__ void k_prep_node(const float* __restrict__ node,
                            unsigned short* __restrict__ node_bf,
                            unsigned short* __restrict__ nodeT_bf) {
    __shared__ float tile[32][33];
    int blk = blockIdx.x;
    int ht = blk & 3;            // 4 tiles of 32 along H
    int it = (blk >> 2) & 31;    // 32 tiles of 32 along N
    int b  = blk >> 7;
    int r = threadIdx.x >> 5;    // 0..7
    int c = threadIdx.x & 31;    // 0..31
    int i0 = it * 32, h0 = ht * 32;
    const float* src = node + ((size_t)b * Ndim + i0) * Hdim + h0;
    for (int rr = r; rr < 32; rr += 8) {
        float v = src[(size_t)rr * Hdim + c];
        tile[rr][c] = v;
        node_bf[((size_t)b * Ndim + i0 + rr) * Hdim + h0 + c] = f2bf(v);
    }
    __syncthreads();
    for (int rr = r; rr < 32; rr += 8)
        nodeT_bf[((size_t)b * Hdim + h0 + rr) * Ndim + i0 + c] = f2bf(tile[c][rr]);
}

// ---------------- K0c: fold W_msg/b_msg through Wu2; zero LN stats ----------
__global__ void k_prep_weights(const float* __restrict__ W_msg, const float* __restrict__ b_msg,
                               const float* __restrict__ W_upd,
                               unsigned short* __restrict__ Wu1_bf,
                               unsigned short* __restrict__ M_bf,
                               float* __restrict__ Ew, float* __restrict__ bbv,
                               float* __restrict__ stats) {
    int t = threadIdx.x;
    stats[t] = 0.f;                                  // 256 floats: sum | sumsq
    for (int e = t; e < Hdim * Hdim; e += 256) {     // Wu1 and M = Wu2 @ W_h
        int o = e >> 7, h = e & 127;
        Wu1_bf[e] = f2bf(W_upd[(size_t)o * 2 * Hdim + h]);
        float s = 0.f;
        for (int k = 0; k < Hdim; ++k)
            s += W_upd[(size_t)o * 2 * Hdim + Hdim + k] * W_msg[(size_t)k * (Hdim + 3) + h];
        M_bf[e] = f2bf(s);
    }
    for (int e = t; e < Hdim * 3; e += 256) {        // E = Wu2 @ W_e
        int o = e / 3, c = e - 3 * o;
        float s = 0.f;
        for (int k = 0; k < Hdim; ++k)
            s += W_upd[(size_t)o * 2 * Hdim + Hdim + k] * W_msg[(size_t)k * (Hdim + 3) + Hdim + c];
        Ew[e] = s;
    }
    for (int o = t; o < Hdim; o += 256) {            // bb = Wu2 @ b_msg
        float s = 0.f;
        for (int k = 0; k < Hdim; ++k)
            s += W_upd[(size_t)o * 2 * Hdim + Hdim + k] * b_msg[k];
        bbv[o] = s;
    }
}

// ---------------- K1: agg[b,j,h] = sum_i mask[j,i]*node[b,i,h]  (WMMA) ------
// Block = 8 waves covering (b, 64 rows) x 128 features; wave w -> col tile w.
// The 64-row A-operand (mask) is shared by all 8 waves: stage it through LDS
// with CDNA5 async-to-LDS DMA, double-buffered (8KB ping-pong), so the next
// 64x64 bf16 chunk streams in while WMMAs consume the current one.
__global__ void k_agg(const unsigned short* __restrict__ mask_bf,
                      const unsigned short* __restrict__ nodeT_bf,
                      unsigned short* __restrict__ agg_bf) {
    __shared__ __align__(16) unsigned short amask[2][64 * 64];   // 2 x 8KB
    int wave = threadIdx.x >> 5;
    int lane = threadIdx.x & 31;
    int rg = blockIdx.x & 15;
    int b  = blockIdx.x >> 4;
    int rowbase = rg * 64;
    int colbase = wave * 16;
    const unsigned short* btbase = nodeT_bf + ((size_t)b * Hdim + colbase) * Ndim;
    unsigned lds_base = (unsigned)(uintptr_t)(&amask[0][0]);

    // Stage one 64x64 bf16 mask chunk (rows rowbase.., K kc..kc+63) into LDS
    // buffer `buf` via async DMA: 512 x 16B units, 2 per thread => ASYNCcnt += 2/wave.
    auto stage = [&](int buf, int kc) {
        int t = threadIdx.x;
#pragma unroll
        for (int rep = 0; rep < 2; ++rep) {
            int u   = t + rep * 256;         // 0..511
            int row = u >> 3;                // 64 rows, 8 x 16B units per row
            int c8  = u & 7;
            const unsigned short* g =
                mask_bf + (size_t)(rowbase + row) * Ndim + kc + c8 * 8;
            unsigned lds = lds_base + (unsigned)buf * 8192u + (unsigned)u * 16u;
            asm volatile("global_load_async_to_lds_b128 %0, %1, off"
                         :: "v"(lds), "v"(g) : "memory");
        }
    };

    v8f acc0 = {}, acc1 = {}, acc2 = {}, acc3 = {};
    stage(0, 0);
    for (int kc = 0; kc < Ndim; kc += 64) {
        int cur = (kc >> 6) & 1;
        if (kc + 64 < Ndim) {
            stage(cur ^ 1, kc + 64);                       // prefetch next chunk
            asm volatile("s_wait_asynccnt 0x2" ::: "memory");  // current chunk done
        } else {
            asm volatile("s_wait_asynccnt 0x0" ::: "memory");
        }
        __syncthreads();                                   // LDS visible to all waves
        const unsigned short* cb = &amask[cur][0];
#pragma unroll
        for (int ks = 0; ks < 64; ks += 32) {
            v16bf bf = load_bt_frag(btbase + kc + ks, Ndim, lane);
            v16bf a0 = load_a_frag(cb +  0 * 64 + ks, 64, lane);
            v16bf a1 = load_a_frag(cb + 16 * 64 + ks, 64, lane);
            v16bf a2 = load_a_frag(cb + 32 * 64 + ks, 64, lane);
            v16bf a3 = load_a_frag(cb + 48 * 64 + ks, 64, lane);
            acc0 = wmma_bf16(a0, bf, acc0);
            acc1 = wmma_bf16(a1, bf, acc1);
            acc2 = wmma_bf16(a2, bf, acc2);
            acc3 = wmma_bf16(a3, bf, acc3);
        }
        __syncthreads();   // all waves done reading before this buffer is re-staged
    }
    int n  = lane & 15;
    int mb = (lane < 16) ? 0 : 8;
    v8f accs[4] = {acc0, acc1, acc2, acc3};
#pragma unroll
    for (int t2 = 0; t2 < 4; ++t2)
#pragma unroll
        for (int r = 0; r < 8; ++r) {
            int j = rowbase + t2 * 16 + mb + r;
            agg_bf[((size_t)b * Ndim + j) * Hdim + colbase + n] = f2bf(accs[t2][r]);
        }
}

// ---------------- K2: e_agg[b,j,c] = sum_i edge[b,i,j,c]*mask[j,i] ----------
// 384 threads = 128 j-values x 3 channels; fully coalesced stream of 384 MB.
__global__ void k_eagg(const float* __restrict__ edge,
                       const unsigned short* __restrict__ maskT_bf,
                       float* __restrict__ eagg) {
    int t  = threadIdx.x;                 // 0..383
    int jb = (blockIdx.x & 7) * 128;
    int b  = blockIdx.x >> 3;
    const float* ebase = edge + (size_t)b * Ndim * Ndim * 3 + (size_t)jb * 3 + t;
    int j = jb + t / 3;
    const unsigned short* mcol = maskT_bf + j;
    const size_t istride = (size_t)Ndim * 3;
    float acc = 0.f;
    for (int i = 0; i < Ndim; i += 4) {
        int ipf = (i + 16 < Ndim) ? (i + 16) : i;
        __builtin_prefetch(ebase + (size_t)ipf * istride, 0, 0);
        float v0 = ebase[(size_t)(i + 0) * istride];
        float v1 = ebase[(size_t)(i + 1) * istride];
        float v2 = ebase[(size_t)(i + 2) * istride];
        float v3 = ebase[(size_t)(i + 3) * istride];
        unsigned short m0 = mcol[(size_t)(i + 0) * Ndim];
        unsigned short m1 = mcol[(size_t)(i + 1) * Ndim];
        unsigned short m2 = mcol[(size_t)(i + 2) * Ndim];
        unsigned short m3 = mcol[(size_t)(i + 3) * Ndim];
        acc += (m0 ? v0 : 0.f) + (m1 ? v1 : 0.f) + (m2 ? v2 : 0.f) + (m3 ? v3 : 0.f);
    }
    eagg[(size_t)b * Ndim * 3 + (size_t)jb * 3 + t] = acc;
}

// ---------------- K3: updated = relu(node@Wu1^T + agg@M^T + eterm) + stats --
__global__ void k_update(const unsigned short* __restrict__ node_bf,
                         const unsigned short* __restrict__ agg_bf,
                         const unsigned short* __restrict__ Wu1_bf,
                         const unsigned short* __restrict__ M_bf,
                         const float* __restrict__ Ew, const float* __restrict__ bbv,
                         const float* __restrict__ deg, const float* __restrict__ eagg,
                         const float* __restrict__ b_upd,
                         float* __restrict__ updated, float* __restrict__ stats) {
    __shared__ float lsum[Hdim];
    __shared__ float lsq[Hdim];
    if (threadIdx.x < Hdim) { lsum[threadIdx.x] = 0.f; lsq[threadIdx.x] = 0.f; }
    __syncthreads();
    int wave = threadIdx.x >> 5;
    int lane = threadIdx.x & 31;
    int rg = blockIdx.x & 15;
    int b  = blockIdx.x >> 4;
    int rowbase = rg * 64;
    int colbase = wave * 16;
    v8f acc0 = {}, acc1 = {}, acc2 = {}, acc3 = {};
    for (int k = 0; k < Hdim; k += 32) {          // node @ Wu1^T
        v16bf bf = load_bt_frag(Wu1_bf + (size_t)colbase * Hdim + k, Hdim, lane);
        v16bf a0 = load_a_frag(node_bf + ((size_t)b * Ndim + rowbase +  0) * Hdim + k, Hdim, lane);
        v16bf a1 = load_a_frag(node_bf + ((size_t)b * Ndim + rowbase + 16) * Hdim + k, Hdim, lane);
        v16bf a2 = load_a_frag(node_bf + ((size_t)b * Ndim + rowbase + 32) * Hdim + k, Hdim, lane);
        v16bf a3 = load_a_frag(node_bf + ((size_t)b * Ndim + rowbase + 48) * Hdim + k, Hdim, lane);
        acc0 = wmma_bf16(a0, bf, acc0);
        acc1 = wmma_bf16(a1, bf, acc1);
        acc2 = wmma_bf16(a2, bf, acc2);
        acc3 = wmma_bf16(a3, bf, acc3);
    }
    for (int k = 0; k < Hdim; k += 32) {          // agg @ (Wu2 W_h)^T
        v16bf bf = load_bt_frag(M_bf + (size_t)colbase * Hdim + k, Hdim, lane);
        v16bf a0 = load_a_frag(agg_bf + ((size_t)b * Ndim + rowbase +  0) * Hdim + k, Hdim, lane);
        v16bf a1 = load_a_frag(agg_bf + ((size_t)b * Ndim + rowbase + 16) * Hdim + k, Hdim, lane);
        v16bf a2 = load_a_frag(agg_bf + ((size_t)b * Ndim + rowbase + 32) * Hdim + k, Hdim, lane);
        v16bf a3 = load_a_frag(agg_bf + ((size_t)b * Ndim + rowbase + 48) * Hdim + k, Hdim, lane);
        acc0 = wmma_bf16(a0, bf, acc0);
        acc1 = wmma_bf16(a1, bf, acc1);
        acc2 = wmma_bf16(a2, bf, acc2);
        acc3 = wmma_bf16(a3, bf, acc3);
    }
    int n  = lane & 15;
    int o  = colbase + n;
    int mb = (lane < 16) ? 0 : 8;
    float e0 = Ew[o * 3 + 0], e1 = Ew[o * 3 + 1], e2 = Ew[o * 3 + 2];
    float bbo = bbv[o], bu = b_upd[o];
    float psum = 0.f, psq = 0.f;
    v8f accs[4] = {acc0, acc1, acc2, acc3};
#pragma unroll
    for (int t2 = 0; t2 < 4; ++t2)
#pragma unroll
        for (int r = 0; r < 8; ++r) {
            int j = rowbase + t2 * 16 + mb + r;
            const float* ea = eagg + ((size_t)b * Ndim + j) * 3;
            float v = accs[t2][r] + ea[0] * e0 + ea[1] * e1 + ea[2] * e2 + deg[j] * bbo + bu;
            v = fmaxf(v, 0.f);
            updated[((size_t)b * Ndim + j) * Hdim + o] = v;
            psum += v;
            psq  += v * v;
        }
    atomicAdd(&lsum[o], psum);
    atomicAdd(&lsq[o],  psq);
    __syncthreads();
    if (threadIdx.x < Hdim)        atomicAdd(&stats[threadIdx.x], lsum[threadIdx.x]);
    else                           atomicAdd(&stats[threadIdx.x], lsq[threadIdx.x - Hdim]);
}

// ---------------- K4: cross-batch per-feature LayerNorm + residual ----------
__global__ void k_norm(const float* __restrict__ updated, const float* __restrict__ node,
                       const float* __restrict__ stats,
                       const float* __restrict__ gamma, const float* __restrict__ beta,
                       float* __restrict__ out) {
    size_t base = ((size_t)blockIdx.x * blockDim.x + threadIdx.x) * 4;
    int o = (int)(base & (Hdim - 1));
    const float inv_cnt = 1.0f / (float)(Bdim * Ndim);
    float4 u  = *(const float4*)(updated + base);
    float4 nd = *(const float4*)(node + base);
    float4 s  = *(const float4*)(stats + o);
    float4 q  = *(const float4*)(stats + Hdim + o);
    float4 g  = *(const float4*)(gamma + o);
    float4 be = *(const float4*)(beta + o);
    const float* up = (const float*)&u;  const float* np = (const float*)&nd;
    const float* sp = (const float*)&s;  const float* qp = (const float*)&q;
    const float* gp = (const float*)&g;  const float* bp = (const float*)&be;
    float4 res;
    float* rp = (float*)&res;
#pragma unroll
    for (int c = 0; c < 4; ++c) {
        float mean = sp[c] * inv_cnt;
        float var  = qp[c] * inv_cnt - mean * mean;
        float inv  = rsqrtf(var + EPS);
        rp[c] = (up[c] - mean) * inv * gp[c] + bp[c] + np[c];
    }
    *(float4*)(out + base) = res;
}

extern "C" void kernel_launch(void* const* d_in, const int* in_sizes, int n_in,
                              void* d_out, int out_size, void* d_ws, size_t ws_size,
                              hipStream_t stream) {
    const float* node  = (const float*)d_in[0];
    const float* edge  = (const float*)d_in[1];
    const int*   adj   = (const int*)d_in[2];
    const float* W_msg = (const float*)d_in[3];
    const float* b_msg = (const float*)d_in[4];
    const float* W_upd = (const float*)d_in[5];
    const float* b_upd = (const float*)d_in[6];
    const float* gamma = (const float*)d_in[7];
    const float* beta  = (const float*)d_in[8];
    float* out = (float*)d_out;

    char* ws = (char*)d_ws;
    size_t off = 0;
    auto take = [&](size_t bytes) -> char* {
        char* p = ws + off;
        off += (bytes + 255) & ~(size_t)255;
        return p;
    };
    unsigned short* mask_bf  = (unsigned short*)take((size_t)Ndim * Ndim * 2);
    unsigned short* maskT_bf = (unsigned short*)take((size_t)Ndim * Ndim * 2);
    unsigned short* node_bf  = (unsigned short*)take((size_t)Bdim * Ndim * Hdim * 2);
    unsigned short* nodeT_bf = (unsigned short*)take((size_t)Bdim * Ndim * Hdim * 2);
    unsigned short* agg_bf   = (unsigned short*)take((size_t)Bdim * Ndim * Hdim * 2);
    unsigned short* Wu1_bf   = (unsigned short*)take((size_t)Hdim * Hdim * 2);
    unsigned short* M_bf     = (unsigned short*)take((size_t)Hdim * Hdim * 2);
    float* Ew    = (float*)take((size_t)Hdim * 3 * 4);
    float* bbv   = (float*)take((size_t)Hdim * 4);
    float* deg   = (float*)take((size_t)Ndim * 4);
    float* eagg  = (float*)take((size_t)Bdim * Ndim * 3 * 4);
    float* upd   = (float*)take((size_t)Bdim * Ndim * Hdim * 4);
    float* stats = (float*)take((size_t)2 * Hdim * 4);

    k_prep_mask<<<Ndim, 256, 0, stream>>>(adj, mask_bf, maskT_bf, deg);
    k_prep_node<<<Bdim * 32 * 4, 256, 0, stream>>>(node, node_bf, nodeT_bf);
    k_prep_weights<<<1, 256, 0, stream>>>(W_msg, b_msg, W_upd, Wu1_bf, M_bf, Ew, bbv, stats);
    k_agg<<<Bdim * 16, 256, 0, stream>>>(mask_bf, nodeT_bf, agg_bf);
    k_eagg<<<Bdim * 8, 384, 0, stream>>>(edge, maskT_bf, eagg);
    k_update<<<Bdim * 16, 256, 0, stream>>>(node_bf, agg_bf, Wu1_bf, M_bf, Ew, bbv, deg,
                                            eagg, b_upd, upd, stats);
    k_norm<<<(Bdim * Ndim * Hdim / 4) / 256, 256, 0, stream>>>(upd, node, stats, gamma, beta, out);
}
